// MP_PDE_Solver_8383776161871
// MI455X (gfx1250) — matrix operand order repaired and checked by reading, not compiled, not run on hardware
//
#include <hip/hip_runtime.h>
#include <math.h>

#define BATCH 2
#define TT    4
#define CCH   3
#define HH    96
#define WWD   96
#define NN    (HH * WWD)   // 9216
#define TCH   12
#define HF    128
#define LL    6
#define KMSG  269
#define KMSGP 288
#define KUPD  256
#define EPSN  1e-5f

typedef __attribute__((ext_vector_type(16))) _Float16 v16h;
typedef __attribute__((ext_vector_type(8)))  _Float16 v8h;
typedef __attribute__((ext_vector_type(8)))  float    v8f;

// swish via hardware approximate reciprocal (v_rcp_f32): the GEMM operands are
// f16 anyway, so an IEEE-exact divide (div_scale/div_fmas expansion) is wasted work.
__device__ __forceinline__ float swishf(float v) {
  return v * __builtin_amdgcn_rcpf(1.0f + __expf(-v));
}

__device__ __forceinline__ v8f wmma32(v16h a, v16h b, v8f c) {
  return __builtin_amdgcn_wmma_f32_16x16x32_f16(false, a, false, b, (short)0, c, false, false);
}

// A-matrix 16x32 f16 half->K mapping: lane m (&15), half p -> K = (p<8?p:p+8) + (lane&16?8:0)
__device__ __forceinline__ int a_koff(int lane, int p) {
  return ((p < 8) ? p : p + 8) + ((lane & 16) ? 8 : 0);
}

__device__ __forceinline__ v16h zero16() {
  v16h z;
#pragma unroll
  for (int i = 0; i < 16; ++i) z[i] = (_Float16)0.f;
  return z;
}

// Load 32 contiguous bytes (16 halves) as a v16h via two 16B vector loads.
__device__ __forceinline__ v16h ld32B(const _Float16* p) {
  v8h lo = *(const v8h*)p;
  v8h hi = *(const v8h*)(p + 8);
  return __builtin_shufflevector(lo, hi, 0, 1, 2, 3, 4, 5, 6, 7, 8, 9, 10, 11, 12, 13, 14, 15);
}

// Swizzled-B fragment: chunk = kc*ntiles + nt; each lane's fragment is contiguous.
__device__ __forceinline__ v16h ldfragB(const _Float16* Wsw, int chunk, int lane) {
  return ld32B(Wsw + ((size_t)chunk * 32 + lane) * 16);
}
// Swizzled-A (hidden tile) fragment: contiguous per lane.
__device__ __forceinline__ v16h ldfragA(const _Float16* Hsw, int kc, int lane) {
  return ld32B(Hsw + ((size_t)kc * 32 + lane) * 16);
}

// A fragment from an f16 row in global memory (row 256B aligned): two 16B loads.
__device__ __forceinline__ v16h ldA_row(const _Float16* row, int lane, int kc) {
  const _Float16* p = row + kc * 32 + ((lane & 16) ? 8 : 0);
  v8h lo = *(const v8h*)p;
  v8h hi = *(const v8h*)(p + 16);
  return __builtin_shufflevector(lo, hi, 0, 1, 2, 3, 4, 5, 6, 7, 8, 9, 10, 11, 12, 13, 14, 15);
}

// B-weight swizzle index for 128-wide outputs: element (k, col) of W[Kpad][128].
__device__ __forceinline__ int bswz(int k, int col) {
  int kc = k >> 5, kl = k & 31;
  int nt = col >> 4, c = col & 15;
  return (((kc * 8 + nt) * 32) + c + ((kl & 8) ? 16 : 0)) * 16 + ((kl & 7) + ((kl & 16) ? 8 : 0));
}
// B-weight swizzle index for 16-wide output (single N tile).
__device__ __forceinline__ int bswz16(int k, int c) {
  int kc = k >> 5, kl = k & 31;
  return ((kc * 32) + c + ((kl & 8) ? 16 : 0)) * 16 + ((kl & 7) + ((kl & 16) ? 8 : 0));
}

// Store C-fragment (bias+swish) into A-swizzled hidden tile (per-wave, 16x128).
__device__ __forceinline__ void store_hidden(_Float16* Hsw, const v8f* acc,
                                             const float* bias, int lane) {
  int cb = lane & 15, rb = (lane & 16) ? 8 : 0;
#pragma unroll
  for (int nt = 0; nt < 8; ++nt) {
    int col = nt * 16 + cb;
    float bv = bias[col];
    int kc = col >> 5, kl = col & 31;
    int base = ((kc * 32) + ((kl & 8) ? 16 : 0) + rb) * 16 + ((kl & 7) + ((kl & 16) ? 8 : 0));
#pragma unroll
    for (int r = 0; r < 8; ++r)
      Hsw[base + r * 16] = (_Float16)swishf(acc[nt][r] + bv);
  }
}

// ---------------------------------------------------------------- k_prep
__global__ void k_prep(const float* __restrict__ x, float* __restrict__ u) {
  int i = blockIdx.x * blockDim.x + threadIdx.x;
  const int total = BATCH * NN * TCH;
  if (i >= total) return;
  int b = i / (NN * TCH);
  int rem = i - b * NN * TCH;
  int n = rem / TCH;
  int k = rem - n * TCH;
  u[i] = x[((size_t)b * TCH + k) * NN + n];
}

// ---------------------------------------------------------------- k_embed
__global__ void k_embed(const float* __restrict__ u, const float* __restrict__ w1,
                        const float* __restrict__ b1, const float* __restrict__ w2,
                        const float* __restrict__ b2, float* __restrict__ hf,
                        _Float16* __restrict__ hfh) {
  extern __shared__ char smem[];
  _Float16* w1s = (_Float16*)smem;            // 32 x 128 swizzled (14 real rows)
  _Float16* w2s = w1s + 32 * HF;              // 128 x 128 swizzled
  float*    b1s = (float*)(w2s + HF * HF);
  float*    b2s = b1s + HF;
  _Float16* hid = (_Float16*)(b2s + HF);      // 4 waves * 16*128 (A-swizzled)

  int tid = threadIdx.x, nthr = blockDim.x;
  for (int i = tid; i < 32 * HF; i += nthr) {
    int r = i >> 7, c = i & 127;
    w1s[bswz(r, c)] = (r < 14) ? (_Float16)w1[r * HF + c] : (_Float16)0.f;
  }
  for (int i = tid; i < HF * HF; i += nthr) {
    int r = i >> 7, c = i & 127;
    w2s[bswz(r, c)] = (_Float16)w2[i];
  }
  for (int i = tid; i < HF; i += nthr) { b1s[i] = b1[i]; b2s[i] = b2[i]; }
  __syncthreads();

  int lane = tid & 31, wave = tid >> 5;
  int tile = blockIdx.x * (blockDim.x >> 5) + wave;
  int b = tile / (NN / 16);
  int nbase = (tile % (NN / 16)) * 16;
  int node = nbase + (lane & 15);
  int yy = node / WWD, xx = node % WWD;
  const float* urow = u + ((size_t)b * NN + node) * TCH;

  const v8f zero = {0.f, 0.f, 0.f, 0.f, 0.f, 0.f, 0.f, 0.f};
  int cb = lane & 15;
  int rb = (lane & 16) ? 8 : 0;

  v8f acc1[8];
#pragma unroll
  for (int nt = 0; nt < 8; ++nt) acc1[nt] = zero;

  v16h a;
#pragma unroll
  for (int p = 0; p < 16; ++p) {
    int k = a_koff(lane, p);
    float v = 0.f;
    if (k < TCH)      v = urow[k];
    else if (k == 12) v = (float)yy * (1.0f / 95.0f);
    else if (k == 13) v = (float)xx * (1.0f / 95.0f);
    a[p] = (_Float16)v;
  }
#pragma unroll
  for (int nt = 0; nt < 8; ++nt) acc1[nt] = wmma32(a, ldfragB(w1s, nt, lane), acc1[nt]);

  _Float16* hw = hid + wave * 16 * HF;
  store_hidden(hw, acc1, b1s, lane);

  v8f acc2[8];
#pragma unroll
  for (int nt = 0; nt < 8; ++nt) acc2[nt] = zero;
#pragma unroll
  for (int kc = 0; kc < 4; ++kc) {
    v16h aa = ldfragA(hw, kc, lane);
#pragma unroll
    for (int nt = 0; nt < 8; ++nt)
      acc2[nt] = wmma32(aa, ldfragB(w2s, kc * 8 + nt, lane), acc2[nt]);
  }
  float*    o  = hf  + ((size_t)b * NN + nbase) * HF;
  _Float16* oh = hfh + ((size_t)b * NN + nbase) * HF;
#pragma unroll
  for (int nt = 0; nt < 8; ++nt) {
    int col = nt * 16 + cb; float bv = b2s[col];
#pragma unroll
    for (int r = 0; r < 8; ++r) {
      float v = swishf(acc2[nt][r] + bv);
      o[(size_t)(r + rb) * HF + col]  = v;
      oh[(size_t)(r + rb) * HF + col] = (_Float16)v;
    }
  }
}

// ---------------------------------------------------------------- k_message
// Gather formulation over the 9 grid offsets (incl. self); analytic degree; no atomics.
__global__ void k_message(const _Float16* __restrict__ hfh, const float* __restrict__ u,
                          const float* __restrict__ w1, const float* __restrict__ b1,
                          const float* __restrict__ w2, const float* __restrict__ b2,
                          _Float16* __restrict__ aggh) {
  extern __shared__ char smem[];
  _Float16* w1s = (_Float16*)smem;            // 288 x 128 swizzled (269 real rows)
  _Float16* w2s = w1s + KMSGP * HF;           // 128 x 128 swizzled
  float*    b1s = (float*)(w2s + HF * HF);
  float*    b2s = b1s + HF;
  _Float16* hid = (_Float16*)(b2s + HF);

  int tid = threadIdx.x, nthr = blockDim.x;
  for (int i = tid; i < KMSGP * HF; i += nthr) {
    int r = i >> 7, c = i & 127;
    w1s[bswz(r, c)] = (r < KMSG) ? (_Float16)w1[r * HF + c] : (_Float16)0.f;
  }
  for (int i = tid; i < HF * HF; i += nthr) {
    int r = i >> 7, c = i & 127;
    w2s[bswz(r, c)] = (_Float16)w2[i];
  }
  for (int i = tid; i < HF; i += nthr) { b1s[i] = b1[i]; b2s[i] = b2[i]; }
  __syncthreads();

  int lane = tid & 31, wave = tid >> 5;
  int tile = blockIdx.x * (blockDim.x >> 5) + wave;
  int b = tile / (NN / 16);
  int nbase = (tile % (NN / 16)) * 16;
  int node = nbase + (lane & 15);
  int ytile = nbase / WWD;          // tile never wraps a grid row (96 = 6*16)
  int xm = node % WWD;
  const _Float16* hfb = hfh + (size_t)b * NN * HF;
  const float*    ub  = u   + (size_t)b * NN * TCH;
  const _Float16* hfDrow = hfb + (size_t)node * HF;
  const float*    uD     = ub  + (size_t)node * TCH;

  int cb = lane & 15;
  int rb = (lane & 16) ? 8 : 0;
  const v8f zero = {0.f, 0.f, 0.f, 0.f, 0.f, 0.f, 0.f, 0.f};
  const v16h zh = zero16();

  // destination-node A fragments are offset-invariant: hoist out of the loop
  v16h aD[4];
#pragma unroll
  for (int kc = 0; kc < 4; ++kc) aD[kc] = ldA_row(hfDrow, lane, kc);

  v8f aggacc[8];
#pragma unroll
  for (int nt = 0; nt < 8; ++nt) aggacc[nt] = zero;

  _Float16* hw = hid + wave * 16 * HF;

  for (int o = 0; o < 9; ++o) {      // (dx,dy) in (-1..1)^2, o==4 is the self loop
    int dx = (o % 3) - 1;
    int dy = (o / 3) - 1;
    int ny = ytile + dy;
    bool nyok = (ny >= 0) && (ny < HH);
    int nx = xm + dx;
    bool valid = nyok && (nx >= 0) && (nx < WWD);
    int src = node + dy * WWD + dx;
    const _Float16* hfSrow = hfb + (size_t)(valid ? src : node) * HF;
    const float*    uS     = ub  + (size_t)(valid ? src : node) * TCH;
    float distv = sqrtf((float)(dx * dx + dy * dy)) * (1.0f / 95.0f);

    // ---- layer 1: [16 x 288] @ [288 x 128]
    v8f acc1[8];
#pragma unroll
    for (int nt = 0; nt < 8; ++nt) acc1[nt] = zero;

#pragma unroll
    for (int kc = 0; kc < 4; ++kc) {           // hf[dst] chunk
#pragma unroll
      for (int nt = 0; nt < 8; ++nt)
        acc1[nt] = wmma32(aD[kc], ldfragB(w1s, kc * 8 + nt, lane), acc1[nt]);
    }
#pragma unroll
    for (int kc = 4; kc < 8; ++kc) {           // hf[src] chunk (masked)
      v16h as = ldA_row(hfSrow, lane, kc - 4);
      as = valid ? as : zh;
#pragma unroll
      for (int nt = 0; nt < 8; ++nt)
        acc1[nt] = wmma32(as, ldfragB(w1s, kc * 8 + nt, lane), acc1[nt]);
    }
    {                                          // chunk 8: u_dst - u_src, dist, zero pad
      v16h a;
#pragma unroll
      for (int p = 0; p < 16; ++p) {
        int koff = a_koff(lane, p);
        float v = 0.f;
        if (koff < TCH)       v = uD[koff] - (valid ? uS[koff] : 0.f);
        else if (koff == TCH) v = distv;
        a[p] = (_Float16)v;
      }
#pragma unroll
      for (int nt = 0; nt < 8; ++nt)
        acc1[nt] = wmma32(a, ldfragB(w1s, 64 + nt, lane), acc1[nt]);
    }

    store_hidden(hw, acc1, b1s, lane);

    // ---- layer 2: [16 x 128] @ [128 x 128]
    v8f acc2[8];
#pragma unroll
    for (int nt = 0; nt < 8; ++nt) acc2[nt] = zero;
#pragma unroll
    for (int kc = 0; kc < 4; ++kc) {
      v16h aa = ldfragA(hw, kc, lane);
#pragma unroll
      for (int nt = 0; nt < 8; ++nt)
        acc2[nt] = wmma32(aa, ldfragB(w2s, kc * 8 + nt, lane), acc2[nt]);
    }
    // masked accumulate (per-row validity)
#pragma unroll
    for (int r = 0; r < 8; ++r) {
      int row = r + rb;
      int nxr = ((nbase + row) % WWD) + dx;
      bool keep = nyok && (nxr >= 0) && (nxr < WWD);
#pragma unroll
      for (int nt = 0; nt < 8; ++nt) {
        float mval = swishf(acc2[nt][r] + b2s[nt * 16 + cb]);
        aggacc[nt][r] += keep ? mval : 0.f;
      }
    }
  }

  // divide by analytic degree and store (f16 mirror only; consumed by k_update)
  _Float16* aggo = aggh + ((size_t)b * NN + nbase) * HF;
#pragma unroll
  for (int r = 0; r < 8; ++r) {
    int row = r + rb;
    int xr = (nbase + row) % WWD;
    int deg = 0;
#pragma unroll
    for (int o = 0; o < 9; ++o) {
      int nx2 = xr + (o % 3) - 1;
      int ny2 = ytile + (o / 3) - 1;
      deg += (nx2 >= 0 && nx2 < WWD && ny2 >= 0 && ny2 < HH) ? 1 : 0;
    }
    float dinv = __builtin_amdgcn_rcpf((float)deg);
#pragma unroll
    for (int nt = 0; nt < 8; ++nt)
      aggo[(size_t)row * HF + nt * 16 + cb] = (_Float16)(aggacc[nt][r] * dinv);
  }
}

// ---------------------------------------------------------------- k_update
__global__ void k_update(const float* __restrict__ hf, const _Float16* __restrict__ hfh,
                         const _Float16* __restrict__ aggh,
                         const float* __restrict__ w1, const float* __restrict__ b1,
                         const float* __restrict__ w2, const float* __restrict__ b2,
                         float* __restrict__ hf2) {
  extern __shared__ char smem[];
  _Float16* w1s = (_Float16*)smem;            // 256 x 128 swizzled
  _Float16* w2s = w1s + KUPD * HF;            // 128 x 128 swizzled
  float*    b1s = (float*)(w2s + HF * HF);
  float*    b2s = b1s + HF;
  _Float16* hid = (_Float16*)(b2s + HF);

  int tid = threadIdx.x, nthr = blockDim.x;
  for (int i = tid; i < KUPD * HF; i += nthr) {
    int r = i >> 7, c = i & 127;
    w1s[bswz(r, c)] = (_Float16)w1[i];
  }
  for (int i = tid; i < HF * HF; i += nthr) {
    int r = i >> 7, c = i & 127;
    w2s[bswz(r, c)] = (_Float16)w2[i];
  }
  for (int i = tid; i < HF; i += nthr) { b1s[i] = b1[i]; b2s[i] = b2[i]; }
  __syncthreads();

  int lane = tid & 31, wave = tid >> 5;
  int tile = blockIdx.x * (blockDim.x >> 5) + wave;
  int b = tile / (NN / 16);
  int nbase = (tile % (NN / 16)) * 16;
  int node = nbase + (lane & 15);
  const _Float16* hfhrow  = hfh  + ((size_t)b * NN + node) * HF;
  const _Float16* agghrow = aggh + ((size_t)b * NN + node) * HF;

  int cb = lane & 15;
  int rb = (lane & 16) ? 8 : 0;
  const v8f zero = {0.f, 0.f, 0.f, 0.f, 0.f, 0.f, 0.f, 0.f};
  _Float16* hw = hid + wave * 16 * HF;

  v8f acc1[8];
#pragma unroll
  for (int nt = 0; nt < 8; ++nt) acc1[nt] = zero;
#pragma unroll
  for (int kc = 0; kc < 4; ++kc) {
    v16h a = ldA_row(hfhrow, lane, kc);
#pragma unroll
    for (int nt = 0; nt < 8; ++nt)
      acc1[nt] = wmma32(a, ldfragB(w1s, kc * 8 + nt, lane), acc1[nt]);
  }
#pragma unroll
  for (int kc = 4; kc < 8; ++kc) {
    v16h a = ldA_row(agghrow, lane, kc - 4);
#pragma unroll
    for (int nt = 0; nt < 8; ++nt)
      acc1[nt] = wmma32(a, ldfragB(w1s, kc * 8 + nt, lane), acc1[nt]);
  }
  store_hidden(hw, acc1, b1s, lane);

  v8f acc2[8];
#pragma unroll
  for (int nt = 0; nt < 8; ++nt) acc2[nt] = zero;
#pragma unroll
  for (int kc = 0; kc < 4; ++kc) {
    v16h aa = ldfragA(hw, kc, lane);
#pragma unroll
    for (int nt = 0; nt < 8; ++nt)
      acc2[nt] = wmma32(aa, ldfragB(w2s, kc * 8 + nt, lane), acc2[nt]);
  }
  // residual: hf2 = hf + swish(layer2 + b2)
  const float* hfrow = hf  + ((size_t)b * NN + nbase) * HF;
  float*       o     = hf2 + ((size_t)b * NN + nbase) * HF;
#pragma unroll
  for (int nt = 0; nt < 8; ++nt) {
    int col = nt * 16 + cb; float bv = b2s[col];
#pragma unroll
    for (int r = 0; r < 8; ++r) {
      size_t idx = (size_t)(r + rb) * HF + col;
      o[idx] = hfrow[idx] + swishf(acc2[nt][r] + bv);
    }
  }
}

// ---------------------------------------------------------------- k_norm
// Instance norm per (batch, feature) over N nodes; writes f32 + f16 mirror.
__global__ void k_norm(const float* __restrict__ src, float* __restrict__ dst,
                       _Float16* __restrict__ dsth) {
  __shared__ float ssum[256];
  __shared__ float ssq[256];
  int bf = blockIdx.x;
  int b = bf / HF, f = bf % HF;
  const float* p = src + (size_t)b * NN * HF + f;
  float s = 0.f, q = 0.f;
  for (int n = threadIdx.x; n < NN; n += blockDim.x) {
    float v = p[(size_t)n * HF];
    s += v; q += v * v;
  }
  ssum[threadIdx.x] = s; ssq[threadIdx.x] = q;
  __syncthreads();
  for (int st = blockDim.x >> 1; st > 0; st >>= 1) {
    if (threadIdx.x < st) {
      ssum[threadIdx.x] += ssum[threadIdx.x + st];
      ssq[threadIdx.x]  += ssq[threadIdx.x + st];
    }
    __syncthreads();
  }
  float mu  = ssum[0] * (1.0f / NN);
  float var = ssq[0] * (1.0f / NN) - mu * mu;
  float inv = __builtin_amdgcn_rsqf(var + EPSN);
  float*    d  = dst  + (size_t)b * NN * HF + f;
  _Float16* dh = dsth + (size_t)b * NN * HF + f;
  for (int n = threadIdx.x; n < NN; n += blockDim.x) {
    float v = (p[(size_t)n * HF] - mu) * inv;
    d[(size_t)n * HF]  = v;
    dh[(size_t)n * HF] = (_Float16)v;
  }
}

// ---------------------------------------------------------------- k_out
__global__ void k_out(const _Float16* __restrict__ hfh, const float* __restrict__ x,
                      const float* __restrict__ w1, const float* __restrict__ b1,
                      const float* __restrict__ w2, const float* __restrict__ b2,
                      float* __restrict__ out) {
  extern __shared__ char smem[];
  _Float16* w1s = (_Float16*)smem;            // 128 x 128 swizzled
  _Float16* w2p = w1s + HF * HF;              // 128 x 16 swizzled (3 real cols)
  float*    b1s = (float*)(w2p + HF * 16);
  _Float16* hid = (_Float16*)(b1s + HF);

  int tid = threadIdx.x, nthr = blockDim.x;
  for (int i = tid; i < HF * HF; i += nthr) {
    int r = i >> 7, c = i & 127;
    w1s[bswz(r, c)] = (_Float16)w1[i];
  }
  for (int i = tid; i < HF * 16; i += nthr) {
    int r = i >> 4, c = i & 15;
    w2p[bswz16(r, c)] = (c < CCH) ? (_Float16)w2[r * CCH + c] : (_Float16)0.f;
  }
  for (int i = tid; i < HF; i += nthr) b1s[i] = b1[i];
  __syncthreads();

  int lane = tid & 31, wave = tid >> 5;
  int tile = blockIdx.x * (blockDim.x >> 5) + wave;
  int b = tile / (NN / 16);
  int nbase = (tile % (NN / 16)) * 16;
  int node = nbase + (lane & 15);
  const _Float16* hfrow = hfh + ((size_t)b * NN + node) * HF;

  int cb = lane & 15;
  int rb = (lane & 16) ? 8 : 0;
  const v8f zero = {0.f, 0.f, 0.f, 0.f, 0.f, 0.f, 0.f, 0.f};
  _Float16* hw = hid + wave * 16 * HF;

  v8f acc1[8];
#pragma unroll
  for (int nt = 0; nt < 8; ++nt) acc1[nt] = zero;
#pragma unroll
  for (int kc = 0; kc < 4; ++kc) {
    v16h a = ldA_row(hfrow, lane, kc);
#pragma unroll
    for (int nt = 0; nt < 8; ++nt)
      acc1[nt] = wmma32(a, ldfragB(w1s, kc * 8 + nt, lane), acc1[nt]);
  }
  store_hidden(hw, acc1, b1s, lane);

  // second layer: single 16-wide N tile (cols 0..2 real), NO swish
  v8f acc2 = zero;
#pragma unroll
  for (int kc = 0; kc < 4; ++kc)
    acc2 = wmma32(ldfragA(hw, kc, lane), ldfragB(w2p, kc, lane), acc2);

  if (cb < CCH) {
    float bv = b2[cb];
#pragma unroll
    for (int r = 0; r < 8; ++r) {
      int n2 = nbase + r + rb;
      float xlast = x[(((size_t)b * TT + (TT - 1)) * CCH + cb) * NN + n2];
      out[((size_t)b * CCH + cb) * NN + n2] = xlast + acc2[r] + bv;
    }
  }
}

// ---------------------------------------------------------------- launch
extern "C" void kernel_launch(void* const* d_in, const int* in_sizes, int n_in,
                              void* d_out, int out_size, void* d_ws, size_t ws_size,
                              hipStream_t stream) {
  (void)in_sizes; (void)n_in; (void)out_size; (void)ws_size;

  const float* x      = (const float*)d_in[0];
  // d_in[1..3]: edge_src / edge_dst / pos — grid structure recomputed analytically.
  const float* emb_w1 = (const float*)d_in[4];
  const float* emb_b1 = (const float*)d_in[5];
  const float* emb_w2 = (const float*)d_in[6];
  const float* emb_b2 = (const float*)d_in[7];
  const float* msg_w1 = (const float*)d_in[8];
  const float* msg_b1 = (const float*)d_in[9];
  const float* msg_w2 = (const float*)d_in[10];
  const float* msg_b2 = (const float*)d_in[11];
  const float* upd_w1 = (const float*)d_in[12];
  const float* upd_b1 = (const float*)d_in[13];
  const float* upd_w2 = (const float*)d_in[14];
  const float* upd_b2 = (const float*)d_in[15];
  const float* out_w1 = (const float*)d_in[16];
  const float* out_b1 = (const float*)d_in[17];
  const float* out_w2 = (const float*)d_in[18];
  const float* out_b2 = (const float*)d_in[19];
  float* out = (float*)d_out;

  // workspace partition (all regions overwritten before read on every call)
  char* ws = (char*)d_ws;
  auto alignup = [](size_t v) { return (v + 255) & ~(size_t)255; };
  float*    u    = (float*)ws;    ws += alignup((size_t)BATCH * NN * TCH * 4);
  float*    hf   = (float*)ws;    ws += alignup((size_t)BATCH * NN * HF * 4);
  float*    hf2  = (float*)ws;    ws += alignup((size_t)BATCH * NN * HF * 4);
  _Float16* hfh  = (_Float16*)ws; ws += alignup((size_t)BATCH * NN * HF * 2);
  _Float16* aggh = (_Float16*)ws;

  const size_t sh_emb = (size_t)(32 * HF + HF * HF) * 2 + 2 * HF * 4 + 4 * 16 * HF * 2;
  const size_t sh_msg = (size_t)(KMSGP * HF + HF * HF) * 2 + 2 * HF * 4 + 4 * 16 * HF * 2;
  const size_t sh_upd = (size_t)(KUPD * HF + HF * HF) * 2 + 2 * HF * 4 + 4 * 16 * HF * 2;
  const size_t sh_out = (size_t)(HF * HF + HF * 16) * 2 + HF * 4 + 4 * 16 * HF * 2;

  (void)hipFuncSetAttribute(reinterpret_cast<const void*>(k_embed),
                            hipFuncAttributeMaxDynamicSharedMemorySize, (int)sh_emb);
  (void)hipFuncSetAttribute(reinterpret_cast<const void*>(k_message),
                            hipFuncAttributeMaxDynamicSharedMemorySize, (int)sh_msg);
  (void)hipFuncSetAttribute(reinterpret_cast<const void*>(k_update),
                            hipFuncAttributeMaxDynamicSharedMemorySize, (int)sh_upd);
  (void)hipFuncSetAttribute(reinterpret_cast<const void*>(k_out),
                            hipFuncAttributeMaxDynamicSharedMemorySize, (int)sh_out);

  const int prep_total = BATCH * NN * TCH;
  k_prep<<<(prep_total + 255) / 256, 256, 0, stream>>>(x, u);

  const int tiles  = BATCH * NN / 16;  // 1152, exact
  const int blocks = tiles / 4;        // 288 blocks x 4 waves x 16-row tiles

  k_embed<<<blocks, 128, sh_emb, stream>>>(u, emb_w1, emb_b1, emb_w2, emb_b2, hf, hfh);

  for (int l = 0; l < LL; ++l) {
    k_message<<<blocks, 128, sh_msg, stream>>>(
        hfh, u, msg_w1 + (size_t)l * KMSG * HF, msg_b1 + (size_t)l * HF,
        msg_w2 + (size_t)l * HF * HF, msg_b2 + (size_t)l * HF, aggh);
    k_update<<<blocks, 128, sh_upd, stream>>>(
        hf, hfh, aggh, upd_w1 + (size_t)l * KUPD * HF, upd_b1 + (size_t)l * HF,
        upd_w2 + (size_t)l * HF * HF, upd_b2 + (size_t)l * HF, hf2);
    k_norm<<<BATCH * HF, 256, 0, stream>>>(hf2, hf, hfh);
  }

  k_out<<<blocks, 128, sh_out, stream>>>(hfh, x, out_w1, out_b1, out_w2, out_b2, out);
}
